// CfCHead_65403761983519
// MI455X (gfx1250) — matrix operand order repaired and checked by reading, not compiled, not run on hardware
//
#include <hip/hip_runtime.h>

typedef __bf16 bf16;
typedef __attribute__((ext_vector_type(16))) __bf16 v16bf;
typedef __attribute__((ext_vector_type(8)))  __bf16 v8bf;
typedef __attribute__((ext_vector_type(8)))  float  v8f;

#define BATCH 65536
// true dims
#define IN0 74
#define H0  269
#define H1  179
#define H2  64
// padded K (input dim rounded up to multiple of 32) per layer
#define K0P 96
#define K1P 288
#define K2P 192
// padded/stored N per layer == next layer's padded K (final layer: 64)
#define N0P 288
#define N1P 192
#define N2P 64

// ---------------------------------------------------------------------------
// Prep: x (f32, [B][74]) -> bf16 zero-padded [B][K0P]
// ---------------------------------------------------------------------------
__global__ void pack_x_kernel(const float* __restrict__ x, bf16* __restrict__ xb) {
    int tid = blockIdx.x * blockDim.x + threadIdx.x;
    if (tid >= BATCH * K0P) return;
    int b = tid / K0P, k = tid - b * K0P;
    float v = (k < IN0) ? x[b * IN0 + k] : 0.0f;
    xb[tid] = (bf16)v;
}

// ---------------------------------------------------------------------------
// Prep: pack one weight matrix (top in_dim rows of the [in+hid, hid] tensor,
// optionally masked) into the per-lane WMMA B-fragment layout:
//   elem index = ((nn*NKT + kk)*32 + lane)*16 + i
//   maps to     W[k = kk*32 + (lane&16) + i][n = nn*16 + (lane&15)]
// Zero-padded outside [in_dim) x [hid).
// ---------------------------------------------------------------------------
__global__ void pack_w_kernel(const float* __restrict__ W, const int* __restrict__ mask,
                              bf16* __restrict__ out, int in_dim, int hid,
                              int Kpad, int Npad) {
    int tid = blockIdx.x * blockDim.x + threadIdx.x;
    if (tid >= Kpad * Npad) return;
    int NKT  = Kpad >> 5;
    int i    = tid & 15;
    int lane = (tid >> 4) & 31;
    int blk  = tid >> 9;              // nn*NKT + kk
    int kk   = blk % NKT;
    int nn   = blk / NKT;
    int n = nn * 16 + (lane & 15);
    int k = kk * 32 + (lane & 16) + i;
    float v = 0.0f;
    if (k < in_dim && n < hid) {
        v = W[k * hid + n];
        if (mask) v *= (float)mask[k * hid + n];
    }
    out[tid] = (bf16)v;
}

// ---------------------------------------------------------------------------
// Fused CfC layer: 4 GEMMs (shared A fragments) + tanh/sigmoid gate epilogue.
// Block = 256 threads = 8 waves; each wave -> one 32x16 output tile
// (2 M-tiles of 16 rows). Block covers 256 batch rows.
// grid = (BATCH/256, Npad/16). NKT is compile-time for full unroll /
// software pipelining of the k loop.
// ---------------------------------------------------------------------------
template <int NKT>
__global__ void __launch_bounds__(256)
cfc_layer_kernel(const bf16* __restrict__ A, int lda,
                 const bf16* __restrict__ W1, const bf16* __restrict__ W2,
                 const bf16* __restrict__ Wa, const bf16* __restrict__ Wb,
                 const float* __restrict__ b1, const float* __restrict__ b2,
                 const float* __restrict__ ba, const float* __restrict__ bb,
                 int trueN,
                 bf16* __restrict__ outB, float* __restrict__ outF, int ldo) {
    const int lane = threadIdx.x & 31;
    const int wave = threadIdx.x >> 5;
    const int m0   = blockIdx.x * 256 + wave * 32;   // first row of M-tile 0
    const int nn   = blockIdx.y;
    const int half = lane >> 4;      // 0: lanes 0-15, 1: lanes 16-31
    const int col  = lane & 15;

    v8f acc1[2] = {{}, {}};
    v8f acc2[2] = {{}, {}};
    v8f acca[2] = {{}, {}};
    v8f accb[2] = {{}, {}};

    const bf16* Arow0 = A + (size_t)(m0 + col) * lda;
    const bf16* Arow1 = Arow0 + (size_t)16 * lda;
    const int   aoff  = half * 16;
    const size_t wbase = ((size_t)nn * NKT) * 512 + (size_t)lane * 16;

#pragma unroll
    for (int kk = 0; kk < NKT; ++kk) {
        // A fragments: rows m0+col and m0+16+col,
        // K = kk*32 + half*16 + {0..15} (contiguous).
        v8bf a0lo = *(const v8bf*)(Arow0 + kk * 32 + aoff);
        v8bf a0hi = *(const v8bf*)(Arow0 + kk * 32 + 8 + aoff);
        v8bf a1lo = *(const v8bf*)(Arow1 + kk * 32 + aoff);
        v8bf a1hi = *(const v8bf*)(Arow1 + kk * 32 + 8 + aoff);
        v16bf a0, a1;
#pragma unroll
        for (int i = 0; i < 8; ++i) {
            a0[i] = a0lo[i]; a0[8 + i] = a0hi[i];
            a1[i] = a1lo[i]; a1[8 + i] = a1hi[i];
        }

        const size_t wb = wbase + (size_t)kk * 512;
        const v8bf* p1 = (const v8bf*)(W1 + wb);
        const v8bf* p2 = (const v8bf*)(W2 + wb);
        const v8bf* pa = (const v8bf*)(Wa + wb);
        const v8bf* pb = (const v8bf*)(Wb + wb);
        v16bf f1, f2, fa, fb;
#pragma unroll
        for (int i = 0; i < 8; ++i) {
            f1[i] = p1[0][i]; f1[8 + i] = p1[1][i];
            f2[i] = p2[0][i]; f2[8 + i] = p2[1][i];
            fa[i] = pa[0][i]; fa[8 + i] = pa[1][i];
            fb[i] = pb[0][i]; fb[8 + i] = pb[1][i];
        }
        acc1[0] = __builtin_amdgcn_wmma_f32_16x16x32_bf16(false, a0, false, f1, (short)0, acc1[0], false, false);
        acc1[1] = __builtin_amdgcn_wmma_f32_16x16x32_bf16(false, a1, false, f1, (short)0, acc1[1], false, false);
        acc2[0] = __builtin_amdgcn_wmma_f32_16x16x32_bf16(false, a0, false, f2, (short)0, acc2[0], false, false);
        acc2[1] = __builtin_amdgcn_wmma_f32_16x16x32_bf16(false, a1, false, f2, (short)0, acc2[1], false, false);
        acca[0] = __builtin_amdgcn_wmma_f32_16x16x32_bf16(false, a0, false, fa, (short)0, acca[0], false, false);
        acca[1] = __builtin_amdgcn_wmma_f32_16x16x32_bf16(false, a1, false, fa, (short)0, acca[1], false, false);
        accb[0] = __builtin_amdgcn_wmma_f32_16x16x32_bf16(false, a0, false, fb, (short)0, accb[0], false, false);
        accb[1] = __builtin_amdgcn_wmma_f32_16x16x32_bf16(false, a1, false, fb, (short)0, accb[1], false, false);
    }

    // Epilogue: CfC closed-form gate. C/D layout: element i -> row
    // (tilebase + half*8 + i), column nn*16 + col.
    const int n = nn * 16 + col;
    const bool valid = n < trueN;
    const float vb1 = valid ? b1[n] : 0.0f;
    const float vb2 = valid ? b2[n] : 0.0f;
    const float vba = valid ? ba[n] : 0.0f;
    const float vbb = valid ? bb[n] : 0.0f;
#pragma unroll
    for (int t = 0; t < 2; ++t) {
        const int mrow = m0 + t * 16 + half * 8;
#pragma unroll
        for (int i = 0; i < 8; ++i) {
            float ff1 = tanhf(acc1[t][i] + vb1);
            float ff2 = tanhf(acc2[t][i] + vb2);
            float z   = (accb[t][i] + vbb) - (acca[t][i] + vba);   // -t_a + t_b
            float tt  = 1.0f / (1.0f + expf(-z));
            float o   = valid ? (ff1 * (1.0f - tt) + tt * ff2) : 0.0f;
            size_t idx = (size_t)(mrow + i) * ldo + n;
            if (outF) outF[idx] = o;
            else      outB[idx] = (bf16)o;
        }
    }
}

// ---------------------------------------------------------------------------
// Host launcher
// ---------------------------------------------------------------------------
extern "C" void kernel_launch(void* const* d_in, const int* in_sizes, int n_in,
                              void* d_out, int out_size, void* d_ws, size_t ws_size,
                              hipStream_t stream) {
    (void)in_sizes; (void)n_in; (void)out_size; (void)ws_size;

    const float* x = (const float*)d_in[0];
    // layer i params start at 1 + i*9:
    // Wf1, bf1, Wf2, bf2, Wta, bta, Wtb, btb, mask
    const float* Wf1[3]; const float* bf1[3];
    const float* Wf2[3]; const float* bf2[3];
    const float* Wta[3]; const float* bta[3];
    const float* Wtb[3]; const float* btb[3];
    const int*   msk[3];
    for (int i = 0; i < 3; ++i) {
        int b = 1 + i * 9;
        Wf1[i] = (const float*)d_in[b + 0]; bf1[i] = (const float*)d_in[b + 1];
        Wf2[i] = (const float*)d_in[b + 2]; bf2[i] = (const float*)d_in[b + 3];
        Wta[i] = (const float*)d_in[b + 4]; bta[i] = (const float*)d_in[b + 5];
        Wtb[i] = (const float*)d_in[b + 6]; btb[i] = (const float*)d_in[b + 7];
        msk[i] = (const int*)  d_in[b + 8];
    }

    // workspace carve-up (256B aligned)
    char* ws = (char*)d_ws;
    size_t off = 0;
    auto carve = [&](size_t bytes) -> void* {
        off = (off + 255) & ~(size_t)255;
        void* p = ws + off;
        off += bytes;
        return p;
    };
    bf16* xb = (bf16*)carve((size_t)BATCH * K0P * sizeof(bf16));
    bf16* o0 = (bf16*)carve((size_t)BATCH * N0P * sizeof(bf16));
    bf16* o1 = (bf16*)carve((size_t)BATCH * N1P * sizeof(bf16));
    bf16* pw[3][4];
    const int KP[3]  = {K0P, K1P, K2P};
    const int NP[3]  = {N0P, N1P, N2P};
    const int IND[3] = {IN0, H0, H1};
    const int HID[3] = {H0, H1, H2};
    for (int l = 0; l < 3; ++l)
        for (int m = 0; m < 4; ++m)
            pw[l][m] = (bf16*)carve((size_t)KP[l] * NP[l] * sizeof(bf16));

    // 1) pack x
    {
        int total = BATCH * K0P;
        pack_x_kernel<<<(total + 255) / 256, 256, 0, stream>>>(x, xb);
    }
    // 2) pack weights (masked for ff1/ff2, dense for time gates)
    for (int l = 0; l < 3; ++l) {
        int total = KP[l] * NP[l];
        int blocks = (total + 255) / 256;
        pack_w_kernel<<<blocks, 256, 0, stream>>>(Wf1[l], msk[l], pw[l][0], IND[l], HID[l], KP[l], NP[l]);
        pack_w_kernel<<<blocks, 256, 0, stream>>>(Wf2[l], msk[l], pw[l][1], IND[l], HID[l], KP[l], NP[l]);
        pack_w_kernel<<<blocks, 256, 0, stream>>>(Wta[l], nullptr, pw[l][2], IND[l], HID[l], KP[l], NP[l]);
        pack_w_kernel<<<blocks, 256, 0, stream>>>(Wtb[l], nullptr, pw[l][3], IND[l], HID[l], KP[l], NP[l]);
    }

    dim3 block(256, 1, 1);
    // 3) layer 0: [B,96] x [96,288] -> o0 bf16 [B,288]   (NKT = 96/32 = 3)
    {
        dim3 grid(BATCH / 256, N0P / 16, 1);
        cfc_layer_kernel<3><<<grid, block, 0, stream>>>(
            xb, K0P, pw[0][0], pw[0][1], pw[0][2], pw[0][3],
            bf1[0], bf2[0], bta[0], btb[0], H0, o0, nullptr, N0P);
    }
    // 4) layer 1: [B,288] x [288,192] -> o1 bf16 [B,192]  (NKT = 288/32 = 9)
    {
        dim3 grid(BATCH / 256, N1P / 16, 1);
        cfc_layer_kernel<9><<<grid, block, 0, stream>>>(
            o0, K1P, pw[1][0], pw[1][1], pw[1][2], pw[1][3],
            bf1[1], bf2[1], bta[1], btb[1], H1, o1, nullptr, N1P);
    }
    // 5) layer 2: [B,192] x [192,64] -> d_out f32 [B,64]  (NKT = 192/32 = 6)
    {
        dim3 grid(BATCH / 256, N2P / 16, 1);
        cfc_layer_kernel<6><<<grid, block, 0, stream>>>(
            o1, K2P, pw[2][0], pw[2][1], pw[2][2], pw[2][3],
            bf1[2], bf2[2], bta[2], btb[2], H2, nullptr, (float*)d_out, N2P);
    }
}
